// CrossAttnMistralBlock_53154515255900
// MI455X (gfx1250) — compile-verified, matrix operand order checked by reading, not run on hardware
//
#include <hip/hip_runtime.h>
#include <math.h>
#include <stdint.h>

// ---------------- problem constants ----------------
constexpr int BB   = 2;
constexpr int QL   = 1024;
constexpr int KVL  = 2048;
constexpr int DIM  = 4096;
constexpr int NH   = 32;
constexpr int NKV  = 8;
constexpr int HD   = 128;
constexpr int INNER = 16384;
constexpr float EPS = 1e-5f;
constexpr float SM_SCALE = 0.08838834764831845f; // 1/sqrt(128)

// ---------------- WMMA vector types ----------------
typedef _Float16 v16h __attribute__((ext_vector_type(16)));
typedef _Float16 v8h  __attribute__((ext_vector_type(8)));
typedef _Float16 v4h  __attribute__((ext_vector_type(4)));
typedef float    v8f  __attribute__((ext_vector_type(8)));

__device__ __forceinline__ v8f wmma_f16(v16h a, v16h b, v8f c) {
  return __builtin_amdgcn_wmma_f32_16x16x32_f16(false, a, false, b, (short)0, c, false, false);
}

// Fragment gather from an LDS row: elems 0-7 = row[lo..lo+7], 8-15 = row[16+lo..16+lo+7]
__device__ __forceinline__ v16h frag_lds(const _Float16* rowp, int lo) {
  v8h a = *(const v8h*)(rowp + lo);
  v8h b = *(const v8h*)(rowp + 16 + lo);
  return __builtin_shufflevector(a, b, 0,1,2,3,4,5,6,7,8,9,10,11,12,13,14,15);
}

// Fragment gather from global fp32 row (convert to f16)
__device__ __forceinline__ v16h frag_global_f32(const float* rowp, int lo) {
  v16h r;
#pragma unroll
  for (int i = 0; i < 8; ++i) {
    r[i]     = (_Float16)rowp[lo + i];
    r[8 + i] = (_Float16)rowp[16 + lo + i];
  }
  return r;
}

// ---------------- CDNA5 async global->LDS DMA (ASYNCcnt tracked) ----------------
// GVS mode: addr = SGPR64 base + per-lane 32-bit VGPR offset. VDST carries the
// LDS byte address (low 32 bits of a generic pointer to __shared__).
__device__ __forceinline__ void async_copy_b128(uint32_t lds_addr,
                                                const _Float16* sbase,
                                                uint32_t voff) {
  asm volatile("global_load_async_to_lds_b128 %0, %1, %2"
               :: "v"(lds_addr), "v"(voff), "s"(sbase)
               : "memory");
}
__device__ __forceinline__ void wait_async0() {
  asm volatile("s_wait_asynccnt 0x0" ::: "memory");
}

// ---------------- fp32 -> f16 conversion ----------------
__global__ __launch_bounds__(256) void cvt_f16_kernel(const float* __restrict__ x,
                                                      _Float16* __restrict__ y, size_t n) {
  size_t i = (size_t)blockIdx.x * 256 + threadIdx.x;
  if (i < n) y[i] = (_Float16)x[i];
}

// ---------------- LayerNorm (fp32 in, f16 out) ----------------
__global__ __launch_bounds__(256) void ln_kernel(const float* __restrict__ x,
                                                 const float* __restrict__ w,
                                                 const float* __restrict__ b,
                                                 _Float16* __restrict__ y) {
  __shared__ float red_s[8], red_s2[8];
  __shared__ float s_mu, s_rstd;
  const int row = blockIdx.x;
  const float* xr = x + (size_t)row * DIM;
  _Float16* yr = y + (size_t)row * DIM;
  const int tid = threadIdx.x;
  float sum = 0.f, sum2 = 0.f;
#pragma unroll
  for (int i = 0; i < DIM / 256; ++i) {
    float v = xr[tid + i * 256];
    sum += v; sum2 += v * v;
  }
#pragma unroll
  for (int d = 1; d < 32; d <<= 1) { sum += __shfl_xor(sum, d); sum2 += __shfl_xor(sum2, d); }
  if ((tid & 31) == 0) { red_s[tid >> 5] = sum; red_s2[tid >> 5] = sum2; }
  __syncthreads();
  if (tid == 0) {
    float s = 0.f, s2 = 0.f;
    for (int i = 0; i < 8; ++i) { s += red_s[i]; s2 += red_s2[i]; }
    float mu = s / DIM;
    float var = s2 / DIM - mu * mu;
    s_mu = mu; s_rstd = rsqrtf(var + EPS);
  }
  __syncthreads();
  const float mu = s_mu, rstd = s_rstd;
#pragma unroll
  for (int i = 0; i < DIM / 256; ++i) {
    int c = tid + i * 256;
    yr[c] = (_Float16)((xr[c] - mu) * rstd * w[c] + b[c]);
  }
}

// ---------------- RoPE (in-place, fp32) ----------------
__global__ __launch_bounds__(256) void rope_kernel(float* __restrict__ x, int seqlen,
                                                   int nheads, int width, size_t total) {
  size_t idx = (size_t)blockIdx.x * 256 + threadIdx.x;
  if (idx >= total) return;
  int j = (int)(idx & 63);
  size_t t = idx >> 6;
  int head = (int)(t % nheads);
  size_t row = t / nheads;
  int pos = (int)(row % seqlen);
  float freq = __powf(10000.f, -(float)j * (1.0f / 64.f));
  float ang = (float)pos * freq;
  float s, c;
  __sincosf(ang, &s, &c);
  float* p = x + row * (size_t)width + (size_t)head * HD;
  float x1 = p[j], x2 = p[j + 64];
  p[j]      = x1 * c - x2 * s;
  p[j + 64] = x2 * c + x1 * s;
}

// ---------------- Async double-buffered WMMA GEMM ----------------
// C[M,N] = A[M,K](f16) @ W[N,K](f16)^T
// MODE 0: Cf = acc (+R residual)      (fp32 out)
// MODE 2: Ch = (f16)( silu(R) * acc ) (fused SwiGLU; R = gate fp32)
// 256 threads = 8 waves; block tile 128x128, K-step 32; wave tile 64x32.
template <int MODE>
__global__ __launch_bounds__(256) void gemm_wmma_async(const _Float16* __restrict__ A,
                                                       const _Float16* __restrict__ W,
                                                       const float* __restrict__ R,
                                                       float* __restrict__ Cf,
                                                       _Float16* __restrict__ Ch,
                                                       int M, int N, int K) {
  __shared__ alignas(16) _Float16 As[2][128][32];
  __shared__ alignas(16) _Float16 Ws[2][128][32];
  const int tid  = threadIdx.x;
  const int lane = tid & 31;
  const int wave = tid >> 5;
  const int waveM = wave >> 2;       // 0..1
  const int waveN = wave & 3;        // 0..3
  const int ln = lane & 15;
  const int lo = (lane >> 4) << 3;   // 0 or 8
  const int rh = lo;
  const int rowBase = blockIdx.y * 128;
  const int colBase = blockIdx.x * 128;

  // per-thread DMA chunk assignment: 512 x 16B chunks per 128x32 f16 tile
  const int c0r = tid >> 2,          c0q = (tid & 3) * 8;
  const int c1r = (tid + 256) >> 2,  c1q = ((tid + 256) & 3) * 8;
  const uint32_t voff0 = (uint32_t)(((size_t)c0r * K + c0q) * sizeof(_Float16));
  const uint32_t voff1 = (uint32_t)(((size_t)c1r * K + c1q) * sizeof(_Float16));
  const _Float16* Abase = A + (size_t)rowBase * K;
  const _Float16* Wbase = W + (size_t)colBase * K;

  auto issue_tile = [&](int buf, int k0) {
    const _Float16* a = Abase + k0;
    const _Float16* w = Wbase + k0;
    async_copy_b128((uint32_t)(uintptr_t)&As[buf][c0r][c0q], a, voff0);
    async_copy_b128((uint32_t)(uintptr_t)&As[buf][c1r][c1q], a, voff1);
    async_copy_b128((uint32_t)(uintptr_t)&Ws[buf][c0r][c0q], w, voff0);
    async_copy_b128((uint32_t)(uintptr_t)&Ws[buf][c1r][c1q], w, voff1);
  };

  v8f acc[4][2];
#pragma unroll
  for (int i = 0; i < 4; ++i)
#pragma unroll
    for (int j = 0; j < 2; ++j)
#pragma unroll
      for (int r = 0; r < 8; ++r) acc[i][j][r] = 0.f;

  const int T = K / 32;
  issue_tile(0, 0);
  for (int it = 0; it < T; ++it) {
    wait_async0();          // my tile `it` copies complete
    __syncthreads();        // everyone's copies complete; prev compute finished
    if (it + 1 < T) issue_tile((it + 1) & 1, (it + 1) * 32);  // overlap DMA w/ WMMA
    const int buf = it & 1;
    v16h fa[4], fb[2];
#pragma unroll
    for (int i = 0; i < 4; ++i) fa[i] = frag_lds(&As[buf][waveM * 64 + i * 16 + ln][0], lo);
#pragma unroll
    for (int j = 0; j < 2; ++j) fb[j] = frag_lds(&Ws[buf][waveN * 32 + j * 16 + ln][0], lo);
#pragma unroll
    for (int i = 0; i < 4; ++i)
#pragma unroll
      for (int j = 0; j < 2; ++j) acc[i][j] = wmma_f16(fa[i], fb[j], acc[i][j]);
  }

#pragma unroll
  for (int i = 0; i < 4; ++i)
#pragma unroll
    for (int j = 0; j < 2; ++j) {
      int col = colBase + waveN * 32 + j * 16 + ln;
      int row0 = rowBase + waveM * 64 + i * 16 + rh;
#pragma unroll
      for (int r = 0; r < 8; ++r) {
        size_t off = (size_t)(row0 + r) * N + col;
        float v = acc[i][j][r];
        if (MODE == 0) {
          if (R) v += R[off];
          Cf[off] = v;
        } else { // MODE 2: fused silu(gate) * up -> f16
          float g = R[off];
          Ch[off] = (_Float16)(g * (1.f / (1.f + __expf(-g))) * v);
        }
      }
    }
}

// ---------------- Flash attention (GQA, mask add, online softmax) ----------------
// grid (QL/64, NH, B), 128 threads = 4 waves, each wave owns 16 q rows; f16 output
__global__ __launch_bounds__(128) void attn_kernel(const float* __restrict__ qb,
                                                   const float* __restrict__ kb,
                                                   const float* __restrict__ vb,
                                                   const float* __restrict__ mask,
                                                   _Float16* __restrict__ ob) {
  __shared__ alignas(16) _Float16 Ks[32][128];   // kv x hd
  __shared__ alignas(16) _Float16 Vt[128][32];   // hd x kv (transposed)
  __shared__ alignas(16) _Float16 Ps[4][16][32]; // per-wave P tile

  const int tid  = threadIdx.x;
  const int lane = tid & 31;
  const int wave = tid >> 5;
  const int ln = lane & 15;
  const int lo = (lane >> 4) << 3;
  const int rh = lo;

  const int qblk = blockIdx.x;
  const int h    = blockIdx.y;
  const int b    = blockIdx.z;
  const int hkv  = h >> 2;

  const int q0 = qblk * 64 + wave * 16;
  const int qrow = q0 + ln;

  v16h fq[4];
  const float* qp = qb + (size_t)(b * QL + qrow) * DIM + (size_t)h * HD;
#pragma unroll
  for (int t = 0; t < 4; ++t) fq[t] = frag_global_f32(qp + t * 32, lo);

  v8f o[8];
#pragma unroll
  for (int j = 0; j < 8; ++j)
#pragma unroll
    for (int r = 0; r < 8; ++r) o[j][r] = 0.f;
  float mrun[8], lrun[8];
#pragma unroll
  for (int r = 0; r < 8; ++r) { mrun[r] = -1e30f; lrun[r] = 0.f; }

  for (int kv0 = 0; kv0 < KVL; kv0 += 32) {
#pragma unroll
    for (int it = 0; it < 8; ++it) {
      int p = tid + it * 128;
      int r = p >> 5;
      int c = (p & 31) * 4;
      size_t goff = (size_t)(b * KVL + kv0 + r) * (NKV * HD) + (size_t)hkv * HD + c;
      float4 kv4 = *(const float4*)(kb + goff);
      v4h hk = { (_Float16)kv4.x, (_Float16)kv4.y, (_Float16)kv4.z, (_Float16)kv4.w };
      *(v4h*)&Ks[r][c] = hk;
      float4 vv4 = *(const float4*)(vb + goff);
      Vt[c + 0][r] = (_Float16)vv4.x;
      Vt[c + 1][r] = (_Float16)vv4.y;
      Vt[c + 2][r] = (_Float16)vv4.z;
      Vt[c + 3][r] = (_Float16)vv4.w;
    }
    __syncthreads();

    v8f s0, s1;
#pragma unroll
    for (int r = 0; r < 8; ++r) { s0[r] = 0.f; s1[r] = 0.f; }
#pragma unroll
    for (int t = 0; t < 4; ++t) {
      v16h fk0 = frag_lds(&Ks[ln][t * 32], lo);
      v16h fk1 = frag_lds(&Ks[16 + ln][t * 32], lo);
      s0 = wmma_f16(fq[t], fk0, s0);
      s1 = wmma_f16(fq[t], fk1, s1);
    }

    const float mv0 = (mask[(size_t)b * KVL + kv0 + ln] != 0.f) ? 1.f : 0.f;
    const float mv1 = (mask[(size_t)b * KVL + kv0 + 16 + ln] != 0.f) ? 1.f : 0.f;

    float p0[8], p1[8], scl[8];
#pragma unroll
    for (int r = 0; r < 8; ++r) {
      float l0 = s0[r] * SM_SCALE + mv0;
      float l1 = s1[r] * SM_SCALE + mv1;
      float mx = fmaxf(l0, l1);
#pragma unroll
      for (int d = 1; d < 16; d <<= 1) mx = fmaxf(mx, __shfl_xor(mx, d));
      float mn = fmaxf(mrun[r], mx);
      p0[r] = __expf(l0 - mn);
      p1[r] = __expf(l1 - mn);
      float sm = p0[r] + p1[r];
#pragma unroll
      for (int d = 1; d < 16; d <<= 1) sm += __shfl_xor(sm, d);
      scl[r] = __expf(mrun[r] - mn);
      lrun[r] = lrun[r] * scl[r] + sm;
      mrun[r] = mn;
    }
#pragma unroll
    for (int j = 0; j < 8; ++j)
#pragma unroll
      for (int r = 0; r < 8; ++r) o[j][r] *= scl[r];

#pragma unroll
    for (int r = 0; r < 8; ++r) {
      int m = rh + r;
      Ps[wave][m][ln]      = (_Float16)p0[r];
      Ps[wave][m][16 + ln] = (_Float16)p1[r];
    }
    v16h fp = frag_lds(&Ps[wave][ln][0], lo);
#pragma unroll
    for (int j = 0; j < 8; ++j) {
      v16h fv = frag_lds(&Vt[j * 16 + ln][0], lo);
      o[j] = wmma_f16(fp, fv, o[j]);
    }
    __syncthreads();
  }

  float inv[8];
#pragma unroll
  for (int r = 0; r < 8; ++r) inv[r] = 1.f / lrun[r];
#pragma unroll
  for (int j = 0; j < 8; ++j)
#pragma unroll
    for (int r = 0; r < 8; ++r) {
      int row = q0 + rh + r;
      ob[(size_t)(b * QL + row) * DIM + (size_t)h * HD + j * 16 + ln] =
          (_Float16)(o[j][r] * inv[r]);
    }
}

// ---------------- launch ----------------
extern "C" void kernel_launch(void* const* d_in, const int* in_sizes, int n_in,
                              void* d_out, int out_size, void* d_ws, size_t ws_size,
                              hipStream_t stream) {
  const float* hidden = (const float*)d_in[0];
  const float* enc    = (const float*)d_in[1];
  const float* mask   = (const float*)d_in[2];
  const float* ln1_w  = (const float*)d_in[3];
  const float* ln1_b  = (const float*)d_in[4];
  const float* q_w    = (const float*)d_in[5];
  const float* k_w    = (const float*)d_in[6];
  const float* v_w    = (const float*)d_in[7];
  const float* o_w    = (const float*)d_in[8];
  const float* ln2_w  = (const float*)d_in[9];
  const float* ln2_b  = (const float*)d_in[10];
  const float* gate_w = (const float*)d_in[11];
  const float* up_w   = (const float*)d_in[12];
  const float* down_w = (const float*)d_in[13];
  float* out = (float*)d_out;

  const size_t nQ  = (size_t)BB * QL * DIM;       // 8,388,608
  const size_t nKVr= (size_t)BB * KVL * NKV * HD; // 4,194,304
  const size_t nE  = (size_t)BB * KVL * DIM;      // 16,777,216
  const size_t nFF = (size_t)BB * QL * INNER;     // 33,554,432
  const size_t nQW = (size_t)DIM * DIM;           // 16,777,216
  const size_t nKW = (size_t)NKV * HD * DIM;      // 4,194,304
  const size_t nGW = (size_t)INNER * DIM;         // 67,108,864

  // fp32 region
  float* ws   = (float*)d_ws;
  float* qbuf = ws;               // q proj (rope'd)
  float* kbuf = qbuf + nQ;
  float* vbuf = kbuf + nKVr;
  float* x1   = vbuf + nKVr;      // residual after attention
  float* gbuf = x1 + nQ;          // gate fp32
  // f16 region
  _Float16* hp   = (_Float16*)(gbuf + nFF);
  _Float16* qw16 = hp;            hp += nQW;
  _Float16* kw16 = hp;            hp += nKW;
  _Float16* vw16 = hp;            hp += nKW;
  _Float16* ow16 = hp;            hp += nQW;
  _Float16* gw16 = hp;            hp += nGW;
  _Float16* uw16 = hp;            hp += nGW;
  _Float16* dw16 = hp;            hp += nGW;
  _Float16* enc16 = hp;           hp += nE;
  _Float16* h1_16 = hp;           hp += nQ;
  _Float16* h2_16 = hp;           hp += nQ;
  _Float16* a16   = hp;           hp += nQ;
  _Float16* act16 = hp;           hp += nFF;

  const int Mq = BB * QL;   // 2048
  const int Mk = BB * KVL;  // 4096

  auto cvt = [&](const float* src, _Float16* dst, size_t n) {
    cvt_f16_kernel<<<(unsigned)((n + 255) / 256), 256, 0, stream>>>(src, dst, n);
  };
  // 0) precision conversion of weights + encoder activations (one pass)
  cvt(q_w, qw16, nQW);  cvt(k_w, kw16, nKW);  cvt(v_w, vw16, nKW);
  cvt(o_w, ow16, nQW);  cvt(gate_w, gw16, nGW); cvt(up_w, uw16, nGW);
  cvt(down_w, dw16, nGW); cvt(enc, enc16, nE);

  // 1) LN1 -> f16
  ln_kernel<<<Mq, 256, 0, stream>>>(hidden, ln1_w, ln1_b, h1_16);
  // 2) Q/K/V projections (async double-buffered WMMA GEMMs), fp32 out for RoPE
  gemm_wmma_async<0><<<dim3(DIM / 128, Mq / 128), 256, 0, stream>>>(
      h1_16, qw16, nullptr, qbuf, nullptr, Mq, DIM, DIM);
  gemm_wmma_async<0><<<dim3((NKV * HD) / 128, Mk / 128), 256, 0, stream>>>(
      enc16, kw16, nullptr, kbuf, nullptr, Mk, NKV * HD, DIM);
  gemm_wmma_async<0><<<dim3((NKV * HD) / 128, Mk / 128), 256, 0, stream>>>(
      enc16, vw16, nullptr, vbuf, nullptr, Mk, NKV * HD, DIM);
  // 3) RoPE in-place
  {
    size_t tq = (size_t)Mq * NH * 64;
    rope_kernel<<<(unsigned)((tq + 255) / 256), 256, 0, stream>>>(qbuf, QL, NH, DIM, tq);
    size_t tk = (size_t)Mk * NKV * 64;
    rope_kernel<<<(unsigned)((tk + 255) / 256), 256, 0, stream>>>(kbuf, KVL, NKV, NKV * HD, tk);
  }
  // 4) flash attention (WMMA), f16 out
  attn_kernel<<<dim3(QL / 64, NH, BB), 128, 0, stream>>>(qbuf, kbuf, vbuf, mask, a16);
  // 5) O projection + residual (fp32 out)
  gemm_wmma_async<0><<<dim3(DIM / 128, Mq / 128), 256, 0, stream>>>(
      a16, ow16, hidden, x1, nullptr, Mq, DIM, DIM);
  // 6) LN2 -> f16
  ln_kernel<<<Mq, 256, 0, stream>>>(x1, ln2_w, ln2_b, h2_16);
  // 7) MLP: gate (fp32), up fused with silu*gate -> f16, down + residual -> d_out
  gemm_wmma_async<0><<<dim3(INNER / 128, Mq / 128), 256, 0, stream>>>(
      h2_16, gw16, nullptr, gbuf, nullptr, Mq, INNER, DIM);
  gemm_wmma_async<2><<<dim3(INNER / 128, Mq / 128), 256, 0, stream>>>(
      h2_16, uw16, gbuf, nullptr, act16, Mq, INNER, DIM);
  gemm_wmma_async<0><<<dim3(DIM / 128, Mq / 128), 256, 0, stream>>>(
      act16, dw16, x1, out, nullptr, Mq, DIM, INNER);
}
